// Hyper_classification_70274254897607
// MI455X (gfx1250) — matrix operand order repaired and checked by reading, not compiled, not run on hardware
//
#include <hip/hip_runtime.h>
#include <hip/hip_bf16.h>

// ---------------------------------------------------------------------------
// Hypergraph GNN forward, CDNA5 (gfx1250) WMMA implementation.
// bf16 activations/weights, f32 WMMA accumulation (v_wmma_f32_16x16x32_bf16),
// fused gather->GEMM->scatter for hyperedge convs, f32 atomics for scatter,
// wave32 shuffle reductions for LN. Multi-M-tile blocks reuse each B fragment
// across several WMMAs to cut L2 weight traffic.
// ---------------------------------------------------------------------------

#define N_NODES 100000
#define D       256
#define LAYERS  4
#define E2      300000
#define E3      200000
#define TGT     2000

typedef __attribute__((ext_vector_type(16))) __bf16 v16bf;
typedef __attribute__((ext_vector_type(8)))  __bf16 v8bf;
typedef __attribute__((ext_vector_type(8)))  float  v8f;

__device__ __forceinline__ __bf16 f2bf(float f) {
    unsigned int u = __float_as_uint(f);
    unsigned int r = (u + 0x7FFFu + ((u >> 16) & 1u)) >> 16;   // RNE
    unsigned short s = (unsigned short)r;
    __bf16 b;
    __builtin_memcpy(&b, &s, 2);
    return b;
}

// A fragment (16x32 bf16, ISA layout): lane<16 -> row=lane, K={kk..kk+7, kk+16..kk+23}
//                                      lane>=16 -> row=lane-16, K shifted by 8.
__device__ __forceinline__ v16bf load_a_frag(const __bf16* tile, int KDIM, int kk, int lane) {
    int row   = lane & 15;
    int abase = (lane < 16) ? 0 : 8;
    const __bf16* p = tile + row * KDIM + kk + abase;
    v8bf lo = *(const v8bf*)(p);
    v8bf hi = *(const v8bf*)(p + 16);
    v16bf r;
#pragma unroll
    for (int i = 0; i < 8; ++i) { r[i] = lo[i]; r[8 + i] = hi[i]; }
    return r;
}

// B fragment (32x16 bf16): lane<16 -> col=lane, K={kk..kk+7, kk+16..kk+23}; lane>=16 -> K shifted by 8.
// WT is N-major: WT[col*KDIM + k] -> two 16B loads per lane.
__device__ __forceinline__ v16bf load_b_frag(const __bf16* WT, int KDIM, int col, int kk, int lane) {
    int kb = (lane < 16) ? 0 : 8;
    const __bf16* p = WT + (size_t)col * KDIM + kk + kb;
    v8bf lo = *(const v8bf*)(p);
    v8bf hi = *(const v8bf*)(p + 16);
    v16bf r;
#pragma unroll
    for (int i = 0; i < 8; ++i) { r[i] = lo[i]; r[8 + i] = hi[i]; }
    return r;
}

__device__ __forceinline__ v8f wmma_bf16(v16bf a, v16bf b, v8f c) {
    return __builtin_amdgcn_wmma_f32_16x16x32_bf16(false, a, false, b, (short)0, c, false, false);
}

// ------------------------- embedding gather (f32 -> bf16) -------------------
__global__ __launch_bounds__(256) void embed_kernel(const int* __restrict__ x,
                                                    const float* __restrict__ emb,
                                                    __bf16* __restrict__ hb) {
    int gid = blockIdx.x * 256 + threadIdx.x;          // over N_NODES*D
    int row = gid >> 8;
    int f   = gid & 255;
    hb[gid] = f2bf(emb[(size_t)x[row] * D + f]);
}

// --------------------- weight convert + transpose to N-major ----------------
__global__ __launch_bounds__(256) void convert_wT(const float* __restrict__ W,
                                                  __bf16* __restrict__ WT,
                                                  int K, int N) {
    int gid = blockIdx.x * 256 + threadIdx.x;          // over K*N
    if (gid >= K * N) return;
    int k = gid / N;
    int n = gid - k * N;
    WT[(size_t)n * K + k] = f2bf(W[gid]);
}

// --------------------------- dense WMMA GEMM (K=N=256) ----------------------
// C[M,256] = A_bf16[gather?(rows)] @ W (WT N-major). Block = 8 waves covering
// 64 rows x 256 cols (4 M-tiles); each wave owns a 64x32 N-slab and reuses
// every B fragment across the 4 M-tiles.
__global__ __launch_bounds__(256) void gemm_wmma_256(const __bf16* __restrict__ Ab,
                                                     const int* __restrict__ gather, int M,
                                                     const __bf16* __restrict__ WT,
                                                     float* __restrict__ C) {
    __shared__ __align__(32) __bf16 sA[64 * 256];
    int tid = threadIdx.x;
    int r0  = blockIdx.x * 64;
    // stage 64x256 bf16 A tile (2048 x uint4); clamp ragged rows (stores guarded)
    for (int c = tid; c < 2048; c += 256) {
        int m = c >> 5, q = c & 31;
        int row = r0 + m;
        if (row >= M) row = M - 1;
        int src = gather ? gather[row] : row;
        ((uint4*)sA)[c] = *((const uint4*)(Ab + (size_t)src * D) + q);
    }
    __syncthreads();
    int wave = tid >> 5, lane = tid & 31;
    v8f acc[2][4];
#pragma unroll
    for (int n = 0; n < 2; ++n)
#pragma unroll
        for (int mt = 0; mt < 4; ++mt) acc[n][mt] = (v8f){0,0,0,0,0,0,0,0};
    int colL = wave * 32 + (lane & 15);
#pragma unroll
    for (int kk = 0; kk < 256; kk += 32) {
        v16bf a[4];
#pragma unroll
        for (int mt = 0; mt < 4; ++mt)
            a[mt] = load_a_frag(sA + mt * 16 * 256, 256, kk, lane);
        v16bf b0 = load_b_frag(WT, 256, colL, kk, lane);
#pragma unroll
        for (int mt = 0; mt < 4; ++mt) acc[0][mt] = wmma_bf16(a[mt], b0, acc[0][mt]);
        v16bf b1 = load_b_frag(WT, 256, colL + 16, kk, lane);
#pragma unroll
        for (int mt = 0; mt < 4; ++mt) acc[1][mt] = wmma_bf16(a[mt], b1, acc[1][mt]);
    }
    int hi = (lane >> 4) << 3;
#pragma unroll
    for (int mt = 0; mt < 4; ++mt)
#pragma unroll
        for (int r = 0; r < 8; ++r) {
            int row = r0 + mt * 16 + r + hi;
            if (row < M) {
                C[(size_t)row * D + colL]      = acc[0][mt][r];
                C[(size_t)row * D + colL + 16] = acc[1][mt][r];
            }
        }
}

// ---------------- fused hyperedge conv: gather -> GEMM -> scatter -----------
// Block handles MT*16 edges. A tile = concat of ARITY node rows staged in LDS;
// 8 waves split N = ARITY*256; each B fragment is reused across MT M-tiles;
// D tiles scatter-added into agg with f32 atomics.
template <int ARITY, int MT>
__global__ __launch_bounds__(256) void edge_conv_wmma(const __bf16* __restrict__ hb,
                                                      const int* __restrict__ eidx, int E,
                                                      const __bf16* __restrict__ WT,
                                                      float* __restrict__ agg) {
    constexpr int KDIM  = ARITY * D;
    constexpr int NTW   = KDIM / 128;                  // N-tiles per wave
    constexpr int MROWS = MT * 16;
    __shared__ __align__(32) __bf16 sA[MROWS * KDIM];
    __shared__ int sIdx[ARITY][MROWS];
    int tid = threadIdx.x;
    int e0  = blockIdx.x * MROWS;
    for (int t = tid; t < ARITY * MROWS; t += 256) {
        int slot = t / MROWS, m = t - slot * MROWS;
        int e = e0 + m;
        if (e >= E) e = E - 1;                         // clamp (scatter guarded)
        sIdx[slot][m] = eidx[(size_t)slot * E + e];
    }
    __syncthreads();
    // stage A tile: MROWS edges x ARITY slots x 32 uint4 (512B per node row)
    for (int c = tid; c < MROWS * ARITY * 32; c += 256) {
        int m    = c / (ARITY * 32);
        int rem  = c - m * (ARITY * 32);
        int slot = rem >> 5;
        int q    = rem & 31;
        const uint4* src = (const uint4*)(hb + (size_t)sIdx[slot][m] * D) + q;
        *((uint4*)(sA + (size_t)m * KDIM + slot * D) + q) = *src;
    }
    __syncthreads();
    int wave = tid >> 5, lane = tid & 31;
    v8f acc[NTW][MT];
#pragma unroll
    for (int nt = 0; nt < NTW; ++nt)
#pragma unroll
        for (int mt = 0; mt < MT; ++mt) acc[nt][mt] = (v8f){0,0,0,0,0,0,0,0};
    for (int kk = 0; kk < KDIM; kk += 32) {
        v16bf a[MT];
#pragma unroll
        for (int mt = 0; mt < MT; ++mt)
            a[mt] = load_a_frag(sA + (size_t)mt * 16 * KDIM, KDIM, kk, lane);
#pragma unroll
        for (int nt = 0; nt < NTW; ++nt) {
            int col = (wave * NTW + nt) * 16 + (lane & 15);
            v16bf b = load_b_frag(WT, KDIM, col, kk, lane);
#pragma unroll
            for (int mt = 0; mt < MT; ++mt)
                acc[nt][mt] = wmma_bf16(a[mt], b, acc[nt][mt]);
        }
    }
    int hi = (lane >> 4) << 3;
#pragma unroll
    for (int nt = 0; nt < NTW; ++nt) {
        int col  = (wave * NTW + nt) * 16 + (lane & 15);
        int slot = col >> 8;                           // which arity position
        int f    = col & 255;                          // feature within D
#pragma unroll
        for (int mt = 0; mt < MT; ++mt)
#pragma unroll
            for (int r = 0; r < 8; ++r) {
                int m = mt * 16 + r + hi;
                if (e0 + m < E) {
                    int node = sIdx[slot][m];
                    __hip_atomic_fetch_add(agg + (size_t)node * D + f, acc[nt][mt][r],
                                           __ATOMIC_RELAXED, __HIP_MEMORY_SCOPE_AGENT);
                }
            }
    }
}

// ------------------- (relu) -> layernorm -> (relu) per row ------------------
// One wave32 per 256-wide row; 8 rows per block.
__global__ __launch_bounds__(256) void ln_kernel(const float* __restrict__ in,
                                                 const float* __restrict__ bias,
                                                 const float* __restrict__ g,
                                                 const float* __restrict__ b,
                                                 int rows, int relu_before, int relu_after,
                                                 __bf16* __restrict__ out_bf,
                                                 float* __restrict__ out_f) {
    int wave = threadIdx.x >> 5, lane = threadIdx.x & 31;
    int row  = blockIdx.x * 8 + wave;
    if (row >= rows) return;
    const float* p = in + (size_t)row * D;
    float v[8], s = 0.f, s2 = 0.f;
#pragma unroll
    for (int i = 0; i < 8; ++i) {
        int c = lane + i * 32;
        float xv = p[c];
        if (bias) xv += bias[c];
        if (relu_before) xv = fmaxf(xv, 0.f);
        v[i] = xv; s += xv; s2 += xv * xv;
    }
#pragma unroll
    for (int off = 16; off > 0; off >>= 1) {
        s  += __shfl_xor(s,  off, 32);
        s2 += __shfl_xor(s2, off, 32);
    }
    float mu   = s * (1.f / D);
    float var  = s2 * (1.f / D) - mu * mu;
    float rstd = rsqrtf(var + 1e-5f);
#pragma unroll
    for (int i = 0; i < 8; ++i) {
        int c = lane + i * 32;
        float y = (v[i] - mu) * rstd * g[c] + b[c];
        if (relu_after) y = fmaxf(y, 0.f);
        if (out_bf) out_bf[(size_t)row * D + c] = f2bf(y);
        if (out_f)  out_f[(size_t)row * D + c]  = y;
    }
}

// --------------------------- final projection to 1 --------------------------
__global__ __launch_bounds__(256) void out_kernel(const float* __restrict__ act,
                                                  const float* __restrict__ w,
                                                  const float* __restrict__ b,
                                                  float* __restrict__ out, int rows) {
    int wave = threadIdx.x >> 5, lane = threadIdx.x & 31;
    int row  = blockIdx.x * 8 + wave;
    if (row >= rows) return;
    float s = 0.f;
#pragma unroll
    for (int i = 0; i < 8; ++i) {
        int c = lane + i * 32;
        s += act[(size_t)row * D + c] * w[c];
    }
#pragma unroll
    for (int off = 16; off > 0; off >>= 1) s += __shfl_xor(s, off, 32);
    if (lane == 0) out[row] = s + b[0];
}

// ---------------------------------------------------------------------------
extern "C" void kernel_launch(void* const* d_in, const int* in_sizes, int n_in,
                              void* d_out, int out_size, void* d_ws, size_t ws_size,
                              hipStream_t stream) {
    const int*   x      = (const int*)d_in[0];
    // d_in[1] = edge_index (unused by reference)
    const int*   e2     = (const int*)d_in[2];
    const int*   e3     = (const int*)d_in[3];
    const int*   tgt    = (const int*)d_in[4];
    const float* emb    = (const float*)d_in[5];
    const float* W_root = (const float*)d_in[6];
    const float* W_e2   = (const float*)d_in[7];
    const float* W_e3   = (const float*)d_in[8];
    const float* ln_g   = (const float*)d_in[9];
    const float* ln_b   = (const float*)d_in[10];
    const float* reg_W  = (const float*)d_in[11];
    const float* reg_b  = (const float*)d_in[12];
    const float* rlng   = (const float*)d_in[13];
    const float* rlnb   = (const float*)d_in[14];
    const float* out_W  = (const float*)d_in[15];
    const float* out_b  = (const float*)d_in[16];
    float*       out    = (float*)d_out;

    // ---- workspace carve (256B aligned) ----
    char* p = (char*)d_ws;
    auto carve = [&](size_t bytes) { char* r = p; p += (bytes + 255) & ~(size_t)255; return r; };
    __bf16* hb      = (__bf16*)carve((size_t)N_NODES * D * 2);   // bf16 activations
    float*  agg     = (float*) carve((size_t)N_NODES * D * 4);   // f32 accumulation
    __bf16* wtroot  = (__bf16*)carve((size_t)D * D * 2);
    __bf16* wte2    = (__bf16*)carve((size_t)(2*D) * (2*D) * 2);
    __bf16* wte3    = (__bf16*)carve((size_t)(3*D) * (3*D) * 2);
    __bf16* wtreg   = (__bf16*)carve((size_t)D * D * 2);
    float*  rlin    = (float*) carve((size_t)TGT * D * 4);
    __bf16* ract_bf = (__bf16*)carve((size_t)TGT * D * 2);
    float*  ract_f  = (float*) carve((size_t)TGT * D * 4);

    // ---- h = emb[x] (bf16) ----
    embed_kernel<<<(N_NODES * D) / 256, 256, 0, stream>>>(x, emb, hb);

    // ---- 4 GNN layers ----
    for (int l = 0; l < LAYERS; ++l) {
        convert_wT<<<(D * D + 255) / 256, 256, 0, stream>>>(W_root + (size_t)l * D * D, wtroot, D, D);
        convert_wT<<<(2*D * 2*D + 255) / 256, 256, 0, stream>>>(W_e2 + (size_t)l * 2*D * 2*D, wte2, 2*D, 2*D);
        convert_wT<<<(3*D * 3*D + 255) / 256, 256, 0, stream>>>(W_e3 + (size_t)l * 3*D * 3*D, wte3, 3*D, 3*D);

        gemm_wmma_256<<<(N_NODES + 63) / 64, 256, 0, stream>>>(hb, nullptr, N_NODES, wtroot, agg);
        edge_conv_wmma<2, 3><<<E2 / 48, 256, 0, stream>>>(hb, e2, E2, wte2, agg);   // 300000/48 = 6250
        edge_conv_wmma<3, 2><<<E3 / 32, 256, 0, stream>>>(hb, e3, E3, wte3, agg);   // 200000/32 = 6250

        ln_kernel<<<N_NODES / 8, 256, 0, stream>>>(agg, nullptr,
                                                   ln_g + (size_t)l * D, ln_b + (size_t)l * D,
                                                   N_NODES, /*relu_before=*/1, /*relu_after=*/0,
                                                   hb, nullptr);                     // h = LN(relu(agg))
    }

    // ---- regression MLP on gathered targets ----
    for (int i = 0; i < 2; ++i) {
        convert_wT<<<(D * D + 255) / 256, 256, 0, stream>>>(reg_W + (size_t)i * D * D, wtreg, D, D);
        gemm_wmma_256<<<(TGT + 63) / 64, 256, 0, stream>>>(i == 0 ? hb : ract_bf,
                                                           i == 0 ? tgt : nullptr, TGT,
                                                           wtreg, rlin);
        ln_kernel<<<TGT / 8, 256, 0, stream>>>(rlin, reg_b + (size_t)i * D,
                                               rlng + (size_t)i * D, rlnb + (size_t)i * D,
                                               TGT, /*relu_before=*/0, /*relu_after=*/1,
                                               ract_bf, ract_f);
    }

    // ---- out = act @ out_W + out_b ----
    out_kernel<<<TGT / 8, 256, 0, stream>>>(ract_f, out_W, out_b, out, TGT);
}
